// SelfAttention_87737591922888
// MI455X (gfx1250) — compile-verified, matrix operand order checked by reading, not run on hardware
//
#include <hip/hip_runtime.h>
#include <hip/hip_bf16.h>

typedef _Float16 half_t;
typedef __attribute__((ext_vector_type(8)))  half_t v8h;
typedef __attribute__((ext_vector_type(16))) half_t v16h;
typedef __attribute__((ext_vector_type(8)))  float  v8f;
typedef __attribute__((ext_vector_type(4)))  int    v4i;

#define AS1 __attribute__((address_space(1)))
#define AS3 __attribute__((address_space(3)))

#define S_LEN 2048
#define DIM   4096
#define NH    32
#define NKV   8
#define HD    128

// ---------------------------------------------------------------------------
// CDNA5 async global->LDS copy (ASYNCcnt) with safe fallback
// ---------------------------------------------------------------------------
#if __has_builtin(__builtin_amdgcn_global_load_async_to_lds_b128) && \
    __has_builtin(__builtin_amdgcn_s_wait_asynccnt)
#define USE_ASYNC_LDS 1
#else
#define USE_ASYNC_LDS 0
#endif

// copy 16 bytes (8 halfs) from global g to LDS l
__device__ __forceinline__ void async_copy16(const half_t* g, half_t* l) {
#if USE_ASYNC_LDS
  __builtin_amdgcn_global_load_async_to_lds_b128((AS1 v4i*)g, (AS3 v4i*)l, 0, 0);
#else
  *(v8h*)l = *(const v8h*)g;
#endif
}

__device__ __forceinline__ void async_wait() {
#if USE_ASYNC_LDS
  __builtin_amdgcn_s_wait_asynccnt(0);
#endif
}

// ---------------------------------------------------------------------------
// WMMA helpers (CDNA5 wave32, v_wmma_f32_16x16x32_f16)
// ---------------------------------------------------------------------------
__device__ __forceinline__ v8f wmma16(v16h a, v16h b, v8f c) {
  return __builtin_amdgcn_wmma_f32_16x16x32_f16(false, a, false, b, (short)0, c,
                                                false, false);
}

__device__ __forceinline__ v16h make_frag(v8h lo, v8h hi) {
  v16h f;
#pragma unroll
  for (int i = 0; i < 8; ++i) { f[i] = lo[i]; f[8 + i] = hi[i]; }
  return f;
}

// Load one 16(row)x32(k) operand fragment from a row-major matrix in LDS.
// ISA layout (16-bit operands): lanes 0-15 -> row = lane, K 0-7 in v0..3 and
// K 16-23 in v4..7; lanes 16-31 -> K 8-15 / 24-31.  Two 16B loads per lane.
__device__ __forceinline__ v16h load_frag(const half_t* base, int ld) {
  const int lane = threadIdx.x & 31;
  const int r = lane & 15;
  const int kb = (lane >> 4) << 3;  // 0 or 8
  v8h lo = *(const v8h*)(base + r * ld + kb);
  v8h hi = *(const v8h*)(base + r * ld + kb + 16);
  return make_frag(lo, hi);
}

// ---------------------------------------------------------------------------
// Elementwise / transpose prep kernels
// ---------------------------------------------------------------------------
__global__ void cvt_kernel(const float* __restrict__ in, half_t* __restrict__ out,
                           int n) {
  int i = blockIdx.x * blockDim.x + threadIdx.x;
  if (i < n) out[i] = (half_t)in[i];
}

// W (K x N, row-major fp32) -> Wt (N x K, row-major f16), 32x32 LDS tiles so
// both the global reads and writes are fully coalesced.
__global__ __launch_bounds__(256) void cvt_tr_kernel(const float* __restrict__ W,
                                                     half_t* __restrict__ Wt,
                                                     int K, int N) {
  __shared__ float tile[32][33];
  const int k0 = blockIdx.y * 32;
  const int n0 = blockIdx.x * 32;
  const int tx = threadIdx.x & 31;
  const int ty = threadIdx.x >> 5;  // 0..7
#pragma unroll
  for (int i = 0; i < 32; i += 8)
    tile[ty + i][tx] = W[(size_t)(k0 + ty + i) * N + n0 + tx];
  __syncthreads();
#pragma unroll
  for (int i = 0; i < 32; i += 8)
    Wt[(size_t)(n0 + ty + i) * K + k0 + tx] = (half_t)tile[tx][ty + i];
}

// In-place RoPE on f16 tensor [S][nheads][128]; freqs fp32 [S][64].
// `scale` folds the attention 1/sqrt(HD) into Q so the flash kernel skips it.
__global__ void rope_kernel(half_t* __restrict__ t, const float* __restrict__ cosT,
                            const float* __restrict__ sinT, int nheads, int total,
                            float scale) {
  int idx = blockIdx.x * blockDim.x + threadIdx.x;
  if (idx >= total) return;
  int i = idx & 63;
  int h = (idx >> 6) % nheads;
  int s = idx / (64 * nheads);
  half_t* p = t + (((size_t)s * nheads + h) << 7) + 2 * i;
  float c = cosT[s * 64 + i], sn = sinT[s * 64 + i];
  float e = (float)p[0], o = (float)p[1];
  p[0] = (half_t)((e * c - o * sn) * scale);
  p[1] = (half_t)((e * sn + o * c) * scale);
}

// V [s][h][d] f16 -> Vt [h][d][s] f16 via 32x32 LDS tiles (coalesced).
__global__ __launch_bounds__(256) void vtr_kernel(const half_t* __restrict__ V,
                                                  half_t* __restrict__ Vt) {
  __shared__ half_t tile[32][33];
  const int h = blockIdx.z;
  const int s0 = blockIdx.x * 32;
  const int d0 = blockIdx.y * 32;
  const int tx = threadIdx.x & 31;
  const int ty = threadIdx.x >> 5;
#pragma unroll
  for (int i = 0; i < 32; i += 8)
    tile[ty + i][tx] = V[((size_t)(s0 + ty + i) * NKV + h) * HD + d0 + tx];
  __syncthreads();
#pragma unroll
  for (int i = 0; i < 32; i += 8)
    Vt[((size_t)h * HD + d0 + ty + i) * S_LEN + s0 + tx] = tile[tx][ty + i];
}

// ---------------------------------------------------------------------------
// Tiled WMMA GEMM:  C(MxN) = A(MxK,f16) @ Bt(NxK,f16)^T, fp32 accumulate.
// 128x128 C tile / block, 8 waves, wave = 2(m) x 4(n) 16x16 tiles, BK=32.
// Double-buffered LDS, async global->LDS prefetch of tile k+1 during tile k.
// ---------------------------------------------------------------------------
template <bool STORE_F32>
__global__ __launch_bounds__(256) void gemm_kernel(
    const half_t* __restrict__ A, const half_t* __restrict__ Bt,
    float* __restrict__ Cf, half_t* __restrict__ Ch, int M, int N, int K) {
  constexpr int LDA = 40;  // 32 + 8 halfs pad (80B rows: 16B aligned)
  __shared__ half_t As[2][128 * LDA];
  __shared__ half_t Bs[2][128 * LDA];

  const int tid = threadIdx.x;
  const int lane = tid & 31;
  const int w = tid >> 5;
  const int wm = w >> 1;  // 0..3 -> 32 rows each
  const int wn = w & 1;   // 0..1 -> 64 cols each
  const int m0 = blockIdx.y * 128;
  const int n0 = blockIdx.x * 128;

  v8f acc[2][4] = {};

  const int lr = tid >> 1;        // 0..127 tile row
  const int lc = (tid & 1) * 16;  // 0/16 k-col base (16 halfs per thread)

  auto stage = [&](int kc, int buf) {
    const half_t* ga = A + (size_t)(m0 + lr) * K + kc + lc;
    async_copy16(ga,     As[buf] + lr * LDA + lc);
    async_copy16(ga + 8, As[buf] + lr * LDA + lc + 8);
    const half_t* gb = Bt + (size_t)(n0 + lr) * K + kc + lc;
    async_copy16(gb,     Bs[buf] + lr * LDA + lc);
    async_copy16(gb + 8, Bs[buf] + lr * LDA + lc + 8);
  };

  stage(0, 0);
  async_wait();
  __syncthreads();

  int buf = 0;
  for (int kc = 0; kc < K; kc += 32, buf ^= 1) {
    if (kc + 32 < K) stage(kc + 32, buf ^ 1);  // async prefetch next tile

    v16h bfr[4];
#pragma unroll
    for (int nt = 0; nt < 4; ++nt)
      bfr[nt] = load_frag(Bs[buf] + (wn * 64 + nt * 16) * LDA, LDA);
#pragma unroll
    for (int mt = 0; mt < 2; ++mt) {
      v16h af = load_frag(As[buf] + (wm * 32 + mt * 16) * LDA, LDA);
#pragma unroll
      for (int nt = 0; nt < 4; ++nt) acc[mt][nt] = wmma16(af, bfr[nt], acc[mt][nt]);
    }
    async_wait();     // prefetched tile landed (this wave)
    __syncthreads();  // ... and for all waves
  }

  // C layout: vgpr v, lanes 0-15 -> M=v, lanes 16-31 -> M=v+8; N = lane&15
  const int col = lane & 15;
  const int rh = (lane >> 4) * 8;
#pragma unroll
  for (int mt = 0; mt < 2; ++mt)
#pragma unroll
    for (int nt = 0; nt < 4; ++nt)
#pragma unroll
      for (int v = 0; v < 8; ++v) {
        const int row = m0 + wm * 32 + mt * 16 + rh + v;
        const int c = n0 + wn * 64 + nt * 16 + col;
        if (STORE_F32)
          Cf[(size_t)row * N + c] = acc[mt][nt][v];
        else
          Ch[(size_t)row * N + c] = (half_t)acc[mt][nt][v];
      }
}

// ---------------------------------------------------------------------------
// Flash attention (no causal mask, matches reference full softmax).
// Block = 128 queries of one head, 8 waves.  Streams t in 64-wide chunks.
// Q fragments (pre-scaled by 1/sqrt(HD)) + O accumulator in registers;
// K/Vt double-buffered in LDS with async prefetch; P staged in LDS.
// ---------------------------------------------------------------------------
__global__ __launch_bounds__(256) void attn_kernel(
    const half_t* __restrict__ Q,   // [S][NH][HD]   (already * 1/sqrt(HD))
    const half_t* __restrict__ Kc,  // [S][NKV][HD]
    const half_t* __restrict__ Vt,  // [NKV][HD][S]
    half_t* __restrict__ O) {       // [S][NH][HD]
  constexpr int LDKK = 136;  // K tile: 64 t-rows x 128 d (+8 pad)
  constexpr int LDT = 72;    // V/P tiles: k = 64 t (+8 pad)
  __shared__ half_t Ks[2][64 * LDKK];
  __shared__ half_t Vs[2][128 * LDT];
  __shared__ half_t Ps[128 * LDT];
  __shared__ float red[2][128];
  __shared__ float row_m[128], row_l[128], row_a[128];

  const int tid = threadIdx.x;
  const int lane = tid & 31;
  const int w = tid >> 5;
  const int wm = w >> 1;  // 0..3 -> 32 q-rows
  const int wn = w & 1;   // 0..1 -> 32 t-cols (scores) / 64 d-cols (output)
  const int h = blockIdx.y;
  const int kvh = h >> 2;
  const int q0 = blockIdx.x * 128;

  if (tid < 128) { row_m[tid] = -1e30f; row_l[tid] = 0.0f; }

  // stage K chunk [64 t][128 d] and Vt chunk [128 d][64 t] into LDS buf
  const int kr = tid >> 2;         // 0..63
  const int kcb = (tid & 3) * 32;  // d base
  const int vr = tid >> 1;         // 0..127 (d)
  const int vcb = (tid & 1) * 32;  // t base
  auto stage = [&](int t0, int buf) {
    const half_t* krow = Kc + ((size_t)(t0 + kr) * NKV + kvh) * HD + kcb;
    half_t* kl = Ks[buf] + kr * LDKK + kcb;
    async_copy16(krow,      kl);
    async_copy16(krow + 8,  kl + 8);
    async_copy16(krow + 16, kl + 16);
    async_copy16(krow + 24, kl + 24);
    const half_t* vrow = Vt + ((size_t)kvh * HD + vr) * S_LEN + t0 + vcb;
    half_t* vl = Vs[buf] + vr * LDT + vcb;
    async_copy16(vrow,      vl);
    async_copy16(vrow + 8,  vl + 8);
    async_copy16(vrow + 16, vl + 16);
    async_copy16(vrow + 24, vl + 24);
  };

  // Q fragments: 2 m-tiles x 4 k-chunks of 32 (head_dim 128), registers.
  v16h qf[2][4];
  {
    const int r = lane & 15;
    const int kb = (lane >> 4) << 3;
#pragma unroll
    for (int mt = 0; mt < 2; ++mt) {
      const half_t* qrow = Q + ((size_t)(q0 + wm * 32 + mt * 16 + r) * NH + h) * HD;
#pragma unroll
      for (int kc = 0; kc < 4; ++kc) {
        v8h lo = *(const v8h*)(qrow + kc * 32 + kb);
        v8h hi = *(const v8h*)(qrow + kc * 32 + kb + 16);
        qf[mt][kc] = make_frag(lo, hi);
      }
    }
  }

  stage(0, 0);
  async_wait();
  __syncthreads();

  v8f oacc[2][4] = {};
  const int col = lane & 15;
  const int rh = (lane >> 4) * 8;

  int buf = 0;
  for (int t0 = 0; t0 < S_LEN; t0 += 64, buf ^= 1) {
    if (t0 + 64 < S_LEN) stage(t0 + 64, buf ^ 1);  // async prefetch next chunk

    // Scores: S = Q @ K^T (contract head_dim; scale pre-folded into Q).
    v8f sacc[2][2] = {};
#pragma unroll
    for (int kc = 0; kc < 4; ++kc) {
      v16h bfr[2];
#pragma unroll
      for (int nt = 0; nt < 2; ++nt)
        bfr[nt] = load_frag(Ks[buf] + (wn * 32 + nt * 16) * LDKK + kc * 32, LDKK);
#pragma unroll
      for (int mt = 0; mt < 2; ++mt)
#pragma unroll
        for (int nt = 0; nt < 2; ++nt)
          sacc[mt][nt] = wmma16(qf[mt][kc], bfr[nt], sacc[mt][nt]);
    }

    // Per-row chunk max: lane-16 shuffle reduce, then cross-wave via LDS.
#pragma unroll
    for (int mt = 0; mt < 2; ++mt)
#pragma unroll
      for (int v = 0; v < 8; ++v) {
        float m = fmaxf(sacc[mt][0][v], sacc[mt][1][v]);
#pragma unroll
        for (int off = 8; off >= 1; off >>= 1) m = fmaxf(m, __shfl_xor(m, off, 16));
        if ((lane & 15) == 0) red[wn][wm * 32 + mt * 16 + rh + v] = m;
      }
    __syncthreads();  // B2
    if (tid < 128) {
      const float cm = fmaxf(red[0][tid], red[1][tid]);
      const float mo = row_m[tid];
      const float mn = fmaxf(mo, cm);
      const float al = __expf(mo - mn);
      row_a[tid] = al;
      row_m[tid] = mn;
      row_l[tid] *= al;
    }
    __syncthreads();  // B3

    // P = exp(s - m) -> LDS; partial row sums; rescale O accumulators.
#pragma unroll
    for (int mt = 0; mt < 2; ++mt)
#pragma unroll
      for (int v = 0; v < 8; ++v) {
        const int row = wm * 32 + mt * 16 + rh + v;
        const float mrow = row_m[row];
        const float al = row_a[row];
        float sum = 0.0f;
#pragma unroll
        for (int nt = 0; nt < 2; ++nt) {
          const float p = __expf(sacc[mt][nt][v] - mrow);
          sum += p;
          Ps[row * LDT + wn * 32 + nt * 16 + col] = (half_t)p;
        }
#pragma unroll
        for (int off = 8; off >= 1; off >>= 1) sum += __shfl_xor(sum, off, 16);
        if ((lane & 15) == 0) red[wn][row] = sum;
#pragma unroll
        for (int nt = 0; nt < 4; ++nt) oacc[mt][nt][v] *= al;
      }
    __syncthreads();  // B4
    if (tid < 128) row_l[tid] += red[0][tid] + red[1][tid];

    // O += P @ V  (contract 64 t; B-operand rows are d from Vt tile).
#pragma unroll
    for (int kc = 0; kc < 2; ++kc) {
      v16h bfr[4];
#pragma unroll
      for (int nt = 0; nt < 4; ++nt)
        bfr[nt] = load_frag(Vs[buf] + (wn * 64 + nt * 16) * LDT + kc * 32, LDT);
#pragma unroll
      for (int mt = 0; mt < 2; ++mt) {
        v16h af = load_frag(Ps + (wm * 32 + mt * 16) * LDT + kc * 32, LDT);
#pragma unroll
        for (int nt = 0; nt < 4; ++nt)
          oacc[mt][nt] = wmma16(af, bfr[nt], oacc[mt][nt]);
      }
    }
    async_wait();     // prefetched K/V chunk landed (this wave)
    __syncthreads();  // B5: all waves; protects Ps/red reuse next chunk
  }

  // Normalize and store O[s][h][d] as f16.
#pragma unroll
  for (int mt = 0; mt < 2; ++mt)
#pragma unroll
    for (int v = 0; v < 8; ++v) {
      const int row = wm * 32 + mt * 16 + rh + v;
      const float inv = 1.0f / row_l[row];
#pragma unroll
      for (int nt = 0; nt < 4; ++nt) {
        const int d = wn * 64 + nt * 16 + col;
        O[((size_t)(q0 + row) * NH + h) * HD + d] = (half_t)(oacc[mt][nt][v] * inv);
      }
    }
}

// ---------------------------------------------------------------------------
// Orchestration
// ---------------------------------------------------------------------------
extern "C" void kernel_launch(void* const* d_in, const int* in_sizes, int n_in,
                              void* d_out, int out_size, void* d_ws, size_t ws_size,
                              hipStream_t stream) {
  (void)in_sizes; (void)n_in; (void)out_size; (void)ws_size;
  const float* x  = (const float*)d_in[0];
  const float* wq = (const float*)d_in[1];
  const float* wk = (const float*)d_in[2];
  const float* wv = (const float*)d_in[3];
  const float* wo = (const float*)d_in[4];
  const float* fc = (const float*)d_in[7];
  const float* fs = (const float*)d_in[8];

  char* ws = (char*)d_ws;
  const size_t MB = 1ull << 20;
  half_t* xh  = (half_t*)(ws + 0 * MB);    // 16 MB  x f16
  half_t* wqt = (half_t*)(ws + 16 * MB);   // 32 MB  wq^T f16
  half_t* wkt = (half_t*)(ws + 48 * MB);   //  8 MB  wk^T f16
  half_t* wvt = (half_t*)(ws + 56 * MB);   //  8 MB  wv^T f16
  half_t* wot = (half_t*)(ws + 64 * MB);   // 32 MB  wo^T f16
  half_t* Qh  = (half_t*)(ws + 96 * MB);   // 16 MB  Q (post-RoPE, pre-scaled)
  half_t* Kh  = (half_t*)(ws + 112 * MB);  //  4 MB  K (post-RoPE)
  half_t* Vh  = (half_t*)(ws + 116 * MB);  //  4 MB  V
  half_t* Vth = (half_t*)(ws + 120 * MB);  //  4 MB  V transposed [h][d][s]
  half_t* Ah  = (half_t*)(ws + 124 * MB);  // 16 MB  attention out

  cvt_kernel<<<(S_LEN * DIM + 255) / 256, 256, 0, stream>>>(x, xh, S_LEN * DIM);
  dim3 gt1(DIM / 32, DIM / 32);
  cvt_tr_kernel<<<gt1, 256, 0, stream>>>(wq, wqt, DIM, DIM);
  dim3 gt2(1024 / 32, DIM / 32);
  cvt_tr_kernel<<<gt2, 256, 0, stream>>>(wk, wkt, DIM, 1024);
  cvt_tr_kernel<<<gt2, 256, 0, stream>>>(wv, wvt, DIM, 1024);
  cvt_tr_kernel<<<gt1, 256, 0, stream>>>(wo, wot, DIM, DIM);

  dim3 gq(DIM / 128, S_LEN / 128);
  gemm_kernel<false><<<gq, 256, 0, stream>>>(xh, wqt, nullptr, Qh, S_LEN, DIM, DIM);
  dim3 gk(1024 / 128, S_LEN / 128);
  gemm_kernel<false><<<gk, 256, 0, stream>>>(xh, wkt, nullptr, Kh, S_LEN, 1024, DIM);
  gemm_kernel<false><<<gk, 256, 0, stream>>>(xh, wvt, nullptr, Vh, S_LEN, 1024, DIM);

  rope_kernel<<<(S_LEN * NH * 64 + 255) / 256, 256, 0, stream>>>(
      Qh, fc, fs, NH, S_LEN * NH * 64, 0.08838834764831845f);  // fold 1/sqrt(HD)
  rope_kernel<<<(S_LEN * NKV * 64 + 255) / 256, 256, 0, stream>>>(
      Kh, fc, fs, NKV, S_LEN * NKV * 64, 1.0f);
  dim3 gv(S_LEN / 32, HD / 32, NKV);
  vtr_kernel<<<gv, 256, 0, stream>>>(Vh, Vth);

  dim3 ga(S_LEN / 128, NH);
  attn_kernel<<<ga, 256, 0, stream>>>(Qh, Kh, Vth, Ah);

  dim3 go(DIM / 128, S_LEN / 128);
  gemm_kernel<true><<<go, 256, 0, stream>>>(Ah, wot, (float*)d_out, nullptr, S_LEN,
                                            DIM, DIM);
}
